// Attention_16277926052536
// MI455X (gfx1250) — compile-verified, matrix operand order checked by reading, not run on hardware
//
#include <hip/hip_runtime.h>

typedef __attribute__((ext_vector_type(16))) _Float16 v16h;
typedef __attribute__((ext_vector_type(8)))  float    v8f;

#define N_NODES 8192
#define IN_FEAT 512
#define OUT_FEAT 256
#define GAT_ALPHA 0.2f

// K-pair base for the 16-bit A/B fragment layout (ISA 7.12.2):
// dword r (0..7), lane-half hl: r<4 -> K=2r+8*hl ; r>=4 -> K=16+2(r-4)+8*hl
__device__ __forceinline__ int kpair_base(int r, int hl) {
    return (r < 4 ? 2 * r : 16 + 2 * (r - 4)) + 8 * hl;
}

union HalfPack { _Float16 h[2]; unsigned u; };
union FragU { uint4 u4[2]; v16h h16; };

// ---------------------------------------------------------------------------
// Kernel 0: pack W (512x256 f32) into f16 B-fragment layout [16 kt][16 nt][32 lane][8 dw]
// and zero src/dst accumulators.  256 blocks x 32 threads.
// ---------------------------------------------------------------------------
__global__ __launch_bounds__(32) void pack_w_kernel(const float* __restrict__ W,
                                                    unsigned* __restrict__ wfrag,
                                                    float* __restrict__ src,
                                                    float* __restrict__ dst) {
    int frag = blockIdx.x;            // 0..255
    int kt = frag >> 4, nt = frag & 15;
    int l = threadIdx.x;              // 0..31
    int hl = l >> 4, c = l & 15;
    int col = nt * 16 + c;

    unsigned out[8];
#pragma unroll
    for (int r = 0; r < 8; ++r) {
        int k = kt * 32 + kpair_base(r, hl);
        HalfPack p;
        p.h[0] = (_Float16)W[k * OUT_FEAT + col];
        p.h[1] = (_Float16)W[(k + 1) * OUT_FEAT + col];
        out[r] = p.u;
    }
    unsigned* base = wfrag + (size_t)frag * 256 + l * 8;
    *(uint4*)(base + 0) = make_uint4(out[0], out[1], out[2], out[3]);
    *(uint4*)(base + 4) = make_uint4(out[4], out[5], out[6], out[7]);

    int gid = frag * 32 + l;          // 0..8191 == N_NODES
    src[gid] = 0.0f;
    dst[gid] = 0.0f;
}

// ---------------------------------------------------------------------------
// Kernel 1: Wh = h @ W via v_wmma_f32_16x16x32_f16 (double-buffered B frags).
// 512 blocks x 256 threads (8 waves); block = 16 rows x 256 cols.
// ---------------------------------------------------------------------------
__global__ __launch_bounds__(256) void wh_kernel(const float* __restrict__ h,
                                                 const unsigned* __restrict__ wfrag,
                                                 const float* __restrict__ a1,
                                                 const float* __restrict__ a2,
                                                 unsigned* __restrict__ whfrag,
                                                 float* __restrict__ src,
                                                 float* __restrict__ dst) {
    __shared__ float lds_h[16 * 516];           // padded stride: conflict-free b128 reads

    int m0 = blockIdx.x * 16;
    int t = threadIdx.x;

    // Stage 16x512 strip of h into LDS (coalesced b128 loads)
    for (int i = t; i < (16 * 512) / 4; i += 256) {
        int row = (i * 4) / 512, col = (i * 4) % 512;
        float4 v = *(const float4*)(h + (size_t)(m0 + row) * IN_FEAT + col);
        *(float4*)(&lds_h[row * 516 + col]) = v;
    }
    __syncthreads();

    int w = t >> 5, l = t & 31;
    int hl = l >> 4, lane16 = l & 15;
    int nu0 = 2 * w, nu1 = 2 * w + 1;

    v8f acc0 = {}, acc1 = {};

    FragU B0[2], B1[2];
    {
        const uint4* bp0 = (const uint4*)(wfrag + ((size_t)nu0 * 256) + l * 8);
        B0[0].u4[0] = bp0[0]; B0[0].u4[1] = bp0[1];
        const uint4* bp1 = (const uint4*)(wfrag + ((size_t)nu1 * 256) + l * 8);
        B1[0].u4[0] = bp1[0]; B1[0].u4[1] = bp1[1];
    }

#pragma unroll 2
    for (int ks = 0; ks < 16; ++ks) {
        int cur = ks & 1, nxt = cur ^ 1;
        // prefetch next iteration's B fragments (latency hidden behind A-build + WMMA)
        if (ks + 1 < 16) {
            const uint4* bp0 =
                (const uint4*)(wfrag + ((size_t)((ks + 1) * 16 + nu0) * 256) + l * 8);
            B0[nxt].u4[0] = bp0[0]; B0[nxt].u4[1] = bp0[1];
            const uint4* bp1 =
                (const uint4*)(wfrag + ((size_t)((ks + 1) * 16 + nu1) * 256) + l * 8);
            B1[nxt].u4[0] = bp1[0]; B1[nxt].u4[1] = bp1[1];
        }

        // A fragment: row = lane16, K slots per layout (two runs of 8 floats)
        int k0 = ks * 32;
        const float* hr = &lds_h[lane16 * 516];
        int b1 = k0 + 8 * hl;
        int b2 = k0 + 16 + 8 * hl;
        float4 f0 = *(const float4*)(hr + b1);
        float4 f1 = *(const float4*)(hr + b1 + 4);
        float4 f2 = *(const float4*)(hr + b2);
        float4 f3 = *(const float4*)(hr + b2 + 4);
        v16h a;
        a[0] = (_Float16)f0.x;  a[1] = (_Float16)f0.y;
        a[2] = (_Float16)f0.z;  a[3] = (_Float16)f0.w;
        a[4] = (_Float16)f1.x;  a[5] = (_Float16)f1.y;
        a[6] = (_Float16)f1.z;  a[7] = (_Float16)f1.w;
        a[8] = (_Float16)f2.x;  a[9] = (_Float16)f2.y;
        a[10] = (_Float16)f2.z; a[11] = (_Float16)f2.w;
        a[12] = (_Float16)f3.x; a[13] = (_Float16)f3.y;
        a[14] = (_Float16)f3.z; a[15] = (_Float16)f3.w;

        acc0 = __builtin_amdgcn_wmma_f32_16x16x32_f16(false, a, false, B0[cur].h16,
                                                      (short)0, acc0, false, false);
        acc1 = __builtin_amdgcn_wmma_f32_16x16x32_f16(false, a, false, B1[cur].h16,
                                                      (short)0, acc1, false, false);
    }

    // --- pack Wh into B-fragment buffer for attention GEMM ---
    // C layout: vgpr r, lane l -> row m0 + r + 8*hl, col n0 + lane16.
    // Pairs (acc[2i], acc[2i+1]) land exactly at frag[kt][nu][l][4*parity + i].
    int kt = blockIdx.x >> 1;
    int parity = blockIdx.x & 1;
    {
        unsigned pk[4];
#pragma unroll
        for (int i = 0; i < 4; ++i) {
            HalfPack p;
            p.h[0] = (_Float16)acc0[2 * i];
            p.h[1] = (_Float16)acc0[2 * i + 1];
            pk[i] = p.u;
        }
        *(uint4*)(whfrag + ((size_t)(kt * 16 + nu0) * 256) + l * 8 + parity * 4) =
            make_uint4(pk[0], pk[1], pk[2], pk[3]);
#pragma unroll
        for (int i = 0; i < 4; ++i) {
            HalfPack p;
            p.h[0] = (_Float16)acc1[2 * i];
            p.h[1] = (_Float16)acc1[2 * i + 1];
            pk[i] = p.u;
        }
        *(uint4*)(whfrag + ((size_t)(kt * 16 + nu1) * 256) + l * 8 + parity * 4) =
            make_uint4(pk[0], pk[1], pk[2], pk[3]);
    }

    // --- src/dst partial reductions ---
    float a1v0 = a1[nu0 * 16 + lane16], a2v0 = a2[nu0 * 16 + lane16];
    float a1v1 = a1[nu1 * 16 + lane16], a2v1 = a2[nu1 * 16 + lane16];
#pragma unroll
    for (int r = 0; r < 8; ++r) {
        float sv = acc0[r] * a1v0 + acc1[r] * a1v1;
        float dv = acc0[r] * a2v0 + acc1[r] * a2v1;
#pragma unroll
        for (int m = 1; m < 16; m <<= 1) {
            sv += __shfl_xor(sv, m, 16);
            dv += __shfl_xor(dv, m, 16);
        }
        if (lane16 == 0) {
            int row = m0 + r + 8 * hl;
            atomicAdd(&src[row], sv);
            atomicAdd(&dst[row], dv);
        }
    }
}

// ---------------------------------------------------------------------------
// Kernel 2: dmax = max(dst)  (safe softmax bound via leaky_relu monotonicity)
// ---------------------------------------------------------------------------
__global__ __launch_bounds__(256) void dmax_kernel(const float* __restrict__ dst,
                                                   float* __restrict__ dmax) {
    __shared__ float red[256];
    float m = -1e30f;
    for (int i = threadIdx.x; i < N_NODES; i += 256) m = fmaxf(m, dst[i]);
    red[threadIdx.x] = m;
    __syncthreads();
    for (int s = 128; s > 0; s >>= 1) {
        if (threadIdx.x < s) red[threadIdx.x] = fmaxf(red[threadIdx.x], red[threadIdx.x + s]);
        __syncthreads();
    }
    if (threadIdx.x == 0) *dmax = red[0];
}

// ---------------------------------------------------------------------------
// Kernel 3: fused masked-softmax(attn) @ Wh + ELU, flash-style single pass.
// 512 blocks x 256 threads; block = 16 rows; K loop over 8192 cols in 32-chunks.
// Software-pipelined: adj/dst prefetched one chunk ahead; B fragments issued
// before the p-computation + barrier so WMMAs rarely wait on loadcnt.
// ---------------------------------------------------------------------------
__global__ __launch_bounds__(256) void attn_kernel(const int* __restrict__ adj,
                                                   const unsigned* __restrict__ whfrag,
                                                   const float* __restrict__ src,
                                                   const float* __restrict__ dst,
                                                   const float* __restrict__ dmaxp,
                                                   float* __restrict__ out) {
    __shared__ unsigned afrag[2][32][8];   // P tile in A-fragment layout (f16 pairs)
    __shared__ float rsum[16];

    int m0 = blockIdx.x * 16;
    int t = threadIdx.x;
    int w = t >> 5, l = t & 31;
    int hl = l >> 4, lane16 = l & 15;
    int nu0 = 2 * w, nu1 = 2 * w + 1;

    // p-producer role: row mrow, column pair index kq (coalesced adj reads)
    int mrow = t >> 4;        // 0..15
    int kq = t & 15;          // 0..15 -> columns j0+2kq, j0+2kq+1
    int alane, adw;           // destination in A-fragment layout
    if (kq < 4)       { alane = mrow;      adw = kq; }
    else if (kq < 8)  { alane = mrow + 16; adw = kq - 4; }
    else if (kq < 12) { alane = mrow;      adw = kq - 4; }
    else              { alane = mrow + 16; adw = kq - 8; }

    float dmax = *dmaxp;
    float srow = src[m0 + mrow];
    float xm = srow + dmax;
    float mi = xm > 0.0f ? xm : GAT_ALPHA * xm;   // >= every e in this row

    const int* adjrow = adj + (size_t)(m0 + mrow) * N_NODES;

    float psum = 0.0f;
    v8f acc0 = {}, acc1 = {};

    // pipeline prologue: chunk 0 adj/dst
    int2   ad = *(const int2*)(adjrow + 2 * kq);
    float2 dj = *(const float2*)(dst + 2 * kq);

    for (int ch = 0; ch < N_NODES / 32; ++ch) {
        // Issue this chunk's B-fragment loads now; latency covered by exp + barrier.
        FragU B0c, B1c;
        const uint4* b0 = (const uint4*)(whfrag + ((size_t)(ch * 16 + nu0) * 256) + l * 8);
        B0c.u4[0] = b0[0]; B0c.u4[1] = b0[1];
        const uint4* b1 = (const uint4*)(whfrag + ((size_t)(ch * 16 + nu1) * 256) + l * 8);
        B1c.u4[0] = b1[0]; B1c.u4[1] = b1[1];

        float e0 = srow + dj.x; e0 = e0 > 0.0f ? e0 : GAT_ALPHA * e0;
        float e1 = srow + dj.y; e1 = e1 > 0.0f ? e1 : GAT_ALPHA * e1;
        float p0 = ad.x > 0 ? __expf(e0 - mi) : 0.0f;
        float p1 = ad.y > 0 ? __expf(e1 - mi) : 0.0f;
        psum += p0 + p1;

        HalfPack pk;
        pk.h[0] = (_Float16)p0;
        pk.h[1] = (_Float16)p1;
        afrag[ch & 1][alane][adw] = pk.u;

        // prefetch next chunk's adj/dst into registers (HBM-critical stream)
        if (ch + 1 < N_NODES / 32) {
            int jn = (ch + 1) * 32 + 2 * kq;
            ad = *(const int2*)(adjrow + jn);
            dj = *(const float2*)(dst + jn);
            __builtin_prefetch(adjrow + (ch + 2) * 32 + 2 * kq, 0, 1);
        }
        __syncthreads();

        FragU A;
        A.u4[0] = *(const uint4*)&afrag[ch & 1][l][0];
        A.u4[1] = *(const uint4*)&afrag[ch & 1][l][4];

        acc0 = __builtin_amdgcn_wmma_f32_16x16x32_f16(false, A.h16, false, B0c.h16,
                                                      (short)0, acc0, false, false);
        acc1 = __builtin_amdgcn_wmma_f32_16x16x32_f16(false, A.h16, false, B1c.h16,
                                                      (short)0, acc1, false, false);
        __syncthreads();
    }

    // --- row sums ---
    if (t < 16) rsum[t] = 0.0f;
    __syncthreads();
    atomicAdd(&rsum[mrow], psum);
    __syncthreads();

    // --- normalize, ELU, store ---
#pragma unroll
    for (int r = 0; r < 8; ++r) {
        int row = r + 8 * hl;
        float inv = 1.0f / rsum[row];
        float v0 = acc0[r] * inv;
        float v1 = acc1[r] * inv;
        v0 = v0 > 0.0f ? v0 : __expf(v0) - 1.0f;
        v1 = v1 > 0.0f ? v1 : __expf(v1) - 1.0f;
        out[(size_t)(m0 + row) * OUT_FEAT + nu0 * 16 + lane16] = v0;
        out[(size_t)(m0 + row) * OUT_FEAT + nu1 * 16 + lane16] = v1;
    }
}

// ---------------------------------------------------------------------------
extern "C" void kernel_launch(void* const* d_in, const int* in_sizes, int n_in,
                              void* d_out, int out_size, void* d_ws, size_t ws_size,
                              hipStream_t stream) {
    const float* h   = (const float*)d_in[0];
    const int*   adj = (const int*)d_in[1];
    const float* W   = (const float*)d_in[2];
    const float* a1  = (const float*)d_in[3];
    const float* a2  = (const float*)d_in[4];
    float* out = (float*)d_out;

    char* ws = (char*)d_ws;
    unsigned* whfrag = (unsigned*)ws;                              // 4 MB: Wh f16 B-fragments
    unsigned* wfrag  = (unsigned*)(ws + (4u << 20));               // 256 KB: W f16 B-fragments
    float* src  = (float*)(ws + (4u << 20) + (256u << 10));        // 32 KB
    float* dstv = src + N_NODES;                                   // 32 KB
    float* dmax = dstv + N_NODES;                                  // 4 B

    pack_w_kernel<<<256, 32, 0, stream>>>(W, wfrag, src, dstv);
    wh_kernel<<<N_NODES / 16, 256, 0, stream>>>(h, wfrag, a1, a2, whfrag, src, dstv);
    dmax_kernel<<<1, 256, 0, stream>>>(dstv, dmax);
    attn_kernel<<<N_NODES / 16, 256, 0, stream>>>(adj, whfrag, src, dstv, dmax, out);
}
